// LACL_48404281426460
// MI455X (gfx1250) — compile-verified, matrix operand order checked by reading, not run on hardware
//
#include <hip/hip_runtime.h>

typedef __bf16 bf16;
typedef __attribute__((ext_vector_type(16))) __bf16 v16bf;
typedef __attribute__((ext_vector_type(8)))  __bf16 v8bf;
typedef __attribute__((ext_vector_type(8)))  float  v8f;

union FragBF { v16bf v; v8bf h[2]; };

// ---------------------------------------------------------------------------
// Templated bf16 WMMA GEMM:  C[M,N] = A[M,K] (row-major) * Bt[N,K]^T
// Block = 256 threads = 8 waves, wave grid 2(M) x 4(N), wave tile 64x64
// (4x4 WMMA tiles -> 16 v_wmma per K-32 step vs 16 b128 fragment loads, 1:1).
// Block tile: 128(M) x 256(N).  M%128==0, N%256==0, K%32==0 required.
// MODE 0: Cf[m*N+n]      = acc                     (f32 store)
// MODE 1: Cb[m*N+n]      = bf16(tanh(acc / *kp))   (knn-mean + tanh)
// MODE 2: Cf[m*realN+n]  = acc      if n < realN   (guarded store)
// MODE 3: Cf[m*realN+n] += acc      if n < realN   (guarded accumulate)
// ---------------------------------------------------------------------------
template <int MODE>
__global__ __launch_bounds__(256) void wmma_gemm_bf16(
    const bf16* __restrict__ A, const bf16* __restrict__ Bt,
    float* __restrict__ Cf, bf16* __restrict__ Cb,
    int M, int N, int K, int realN, const int* __restrict__ kp)
{
    const int lane  = threadIdx.x & 31;
    const int wave  = threadIdx.x >> 5;       // 0..7
    const int waveM = wave >> 2;              // 0..1
    const int waveN = wave & 3;               // 0..3
    const int m0 = blockIdx.y * 128 + waveM * 64;
    const int n0 = blockIdx.x * 256 + waveN * 64;

    const int row16 = lane & 15;              // M index for A frag / N index for B frag
    const int khalf = lane >> 4;              // which K half-group this lane holds

    // per-lane base pointers (two 16B chunks per fragment at +0 and +16 elems)
    const bf16* ap[4];
    const bf16* bp[4];
#pragma unroll
    for (int i = 0; i < 4; ++i)
        ap[i] = A + (size_t)(m0 + i * 16 + row16) * K + khalf * 8;
#pragma unroll
    for (int j = 0; j < 4; ++j)
        bp[j] = Bt + (size_t)(n0 + j * 16 + row16) * K + khalf * 8;

    v8f acc[4][4] = {};

    for (int k0 = 0; k0 < K; k0 += 32) {
        FragBF fa[4], fb[4];
#pragma unroll
        for (int i = 0; i < 4; ++i) {
            fa[i].h[0] = *(const v8bf*)(ap[i] + k0);
            fa[i].h[1] = *(const v8bf*)(ap[i] + k0 + 16);
        }
#pragma unroll
        for (int j = 0; j < 4; ++j) {
            fb[j].h[0] = *(const v8bf*)(bp[j] + k0);
            fb[j].h[1] = *(const v8bf*)(bp[j] + k0 + 16);
        }
#pragma unroll
        for (int i = 0; i < 4; ++i)
#pragma unroll
            for (int j = 0; j < 4; ++j)
                acc[i][j] = __builtin_amdgcn_wmma_f32_16x16x32_bf16(
                    false, fa[i].v, false, fb[j].v,
                    (short)0, acc[i][j], false, false);
    }

    float inv_k = 1.0f;
    if constexpr (MODE == 1) inv_k = 1.0f / (float)kp[0];

    // C layout: lane holds column n = lane&15; VGPR r holds row r + 8*(lane>>4)
#pragma unroll
    for (int i = 0; i < 4; ++i)
#pragma unroll
        for (int j = 0; j < 4; ++j)
#pragma unroll
            for (int r = 0; r < 8; ++r) {
                const int cm = m0 + i * 16 + khalf * 8 + r;
                const int cn = n0 + j * 16 + row16;
                const float v = acc[i][j][r];
                if constexpr (MODE == 0) {
                    Cf[(size_t)cm * N + cn] = v;
                } else if constexpr (MODE == 1) {
                    Cb[(size_t)cm * N + cn] = (bf16)tanhf(v * inv_k);
                } else if constexpr (MODE == 2) {
                    if (cn < realN) Cf[(size_t)cm * realN + cn] = v;
                } else {
                    if (cn < realN) Cf[(size_t)cm * realN + cn] += v;
                }
            }
}

// ---------------------------------------------------------------------------
// Elementwise f32 -> bf16
// ---------------------------------------------------------------------------
__global__ void cvt_f32_bf16(const float* __restrict__ in, bf16* __restrict__ out,
                             size_t n)
{
    size_t i = (size_t)blockIdx.x * blockDim.x + threadIdx.x;
    if (i < n) out[i] = (bf16)in[i];
}

// Transpose + convert + zero-pad:  out[n*K + k] = bf16(in[k*Nsrc + n]), n<Nsrc else 0
__global__ void cvt_transpose_pad(const float* __restrict__ in, bf16* __restrict__ out,
                                  int K, int Nsrc, int Npad)
{
    size_t i = (size_t)blockIdx.x * blockDim.x + threadIdx.x;
    size_t total = (size_t)Npad * K;
    if (i >= total) return;
    int k = (int)(i % K);
    int n = (int)(i / K);
    float v = (n < Nsrc) ? in[(size_t)k * Nsrc + n] : 0.0f;
    out[i] = (bf16)v;
}

// ---------------------------------------------------------------------------
// Per-row reciprocal L2 norm of Xe [M, D]
// ---------------------------------------------------------------------------
__global__ __launch_bounds__(256) void row_rnorm(const float* __restrict__ Xe,
                                                 float* __restrict__ rn, int D)
{
    __shared__ float red[256];
    const int row = blockIdx.x;
    float s = 0.0f;
    for (int j = threadIdx.x; j < D; j += 256) {
        float v = Xe[(size_t)row * D + j];
        s += v * v;
    }
    red[threadIdx.x] = s;
    __syncthreads();
    for (int st = 128; st > 0; st >>= 1) {
        if (threadIdx.x < st) red[threadIdx.x] += red[threadIdx.x + st];
        __syncthreads();
    }
    if (threadIdx.x == 0) rn[row] = rsqrtf(fmaxf(red[0], 1e-30f));
}

// Fused prep: relu(Xe)->bf16, Xe*rnorm->bf16 (Xn), Xe^T->bf16 (for A_adj @ Xe GEMM)
__global__ void prep_pass(const float* __restrict__ Xe, const float* __restrict__ rn,
                          bf16* __restrict__ reluB, bf16* __restrict__ xnB,
                          bf16* __restrict__ xeT, int M, int D)
{
    size_t i = (size_t)blockIdx.x * blockDim.x + threadIdx.x;
    if (i >= (size_t)M * D) return;
    int d = (int)(i % D);
    int m = (int)(i / D);
    float v = Xe[i];
    reluB[i] = (bf16)fmaxf(v, 0.0f);
    xnB[i]   = (bf16)(v * rn[m]);
    xeT[(size_t)d * M + m] = (bf16)v;
}

// ---------------------------------------------------------------------------
// Exact per-row top-k via nibble radix-select on monotone float keys.
// One block per row; N columns; writes bf16 adjacency (1.0 for selected).
// Self similarity (j==row) excluded by forcing its key to 0 (minimum).
// ---------------------------------------------------------------------------
__global__ __launch_bounds__(256) void topk_adjacency(const float* __restrict__ S,
                                                      bf16* __restrict__ Adj,
                                                      int N, const int* __restrict__ kp)
{
    __shared__ unsigned keys[4096];
    __shared__ unsigned hist[16];
    __shared__ unsigned s_prefix, s_remaining;
    __shared__ int s_quota;

    const int row = blockIdx.x;
    const int k   = kp[0];
    const float* srow = S + (size_t)row * N;

    for (int j = threadIdx.x; j < N; j += 256) {
        unsigned u = __float_as_uint(srow[j]);
        u = (u & 0x80000000u) ? ~u : (u | 0x80000000u);   // order-preserving map
        if (j == row) u = 0u;                              // exclude self
        keys[j] = u;
    }
    __syncthreads();

    unsigned prefix = 0, remaining = (unsigned)k;
    for (int shift = 28; shift >= 0; shift -= 4) {
        if (threadIdx.x < 16) hist[threadIdx.x] = 0;
        __syncthreads();
        const unsigned mask = (shift == 28) ? 0u : (0xFFFFFFFFu << (shift + 4));
        for (int j = threadIdx.x; j < N; j += 256) {
            unsigned u = keys[j];
            if ((u & mask) == prefix)
                atomicAdd(&hist[(u >> shift) & 15u], 1u);
        }
        __syncthreads();
        if (threadIdx.x == 0) {
            unsigned rem = remaining;
            unsigned d = 15;
            for (; d > 0; --d) {
                if (hist[d] >= rem) break;
                rem -= hist[d];
            }
            s_prefix    = prefix | (d << shift);
            s_remaining = rem;
        }
        __syncthreads();
        prefix    = s_prefix;
        remaining = s_remaining;
        __syncthreads();
    }

    if (threadIdx.x == 0) s_quota = (int)remaining;   // ties at threshold to take
    __syncthreads();
    const unsigned T = prefix;                         // exact k-th largest key

    bf16* arow = Adj + (size_t)row * N;
    for (int j = threadIdx.x; j < N; j += 256) {
        unsigned u = keys[j];
        float val = 0.0f;
        if (u > T) {
            val = 1.0f;
        } else if (u == T) {
            if (atomicSub(&s_quota, 1) > 0) val = 1.0f;
        }
        arow[j] = (bf16)val;
    }
}

// ---------------------------------------------------------------------------
// Host-side orchestration
// ---------------------------------------------------------------------------
extern "C" void kernel_launch(void* const* d_in, const int* in_sizes, int n_in,
                              void* d_out, int out_size, void* d_ws, size_t ws_size,
                              hipStream_t stream)
{
    (void)in_sizes; (void)n_in; (void)out_size; (void)ws_size;

    constexpr int M  = 4096;   // N_SAMPLES
    constexpr int K1 = 512;    // BACKBONE_OUT
    constexpr int D  = 8192;   // EXPANSION
    constexpr int NC = 1000;   // NUM_CLASSES
    constexpr int NCp = 1024;  // padded classes

    const float* X  = (const float*)d_in[0];
    const float* Wb = (const float*)d_in[1];
    const float* Wm = (const float*)d_in[2];
    const float* Wc = (const float*)d_in[3];
    const int*   kp = (const int*)d_in[4];
    float* out = (float*)d_out;

    char* w = (char*)d_ws;
    auto alloc = [&](size_t bytes) -> void* {
        void* p = (void*)w;
        w += (bytes + 255) & ~(size_t)255;
        return p;
    };
    bf16*  Xb   = (bf16*) alloc((size_t)M * K1 * 2);      //   4 MB  X bf16
    bf16*  WbT  = (bf16*) alloc((size_t)D * K1 * 2);      //   8 MB  W_buffer^T
    bf16*  WmT  = (bf16*) alloc((size_t)NCp * D * 2);     //  16 MB  W_main^T (padded)
    bf16*  WcT  = (bf16*) alloc((size_t)NCp * D * 2);     //  16 MB  W_comp^T (padded)
    float* Xe   = (float*)alloc((size_t)M * D * 4);       // 128 MB  Xe f32
    bf16*  reluB= (bf16*) alloc((size_t)M * D * 2);       //  64 MB  relu(Xe)
    bf16*  xnB  = (bf16*) alloc((size_t)M * D * 2);       //  64 MB  Xn (reused as Xc)
    bf16*  xeT  = (bf16*) alloc((size_t)D * M * 2);       //  64 MB  Xe^T
    float* rn   = (float*)alloc((size_t)M * 4);           //  16 KB  1/||row||
    float* Smat = Xe;                                     // reuse: sim matrix (64 MB)
    bf16*  Adj  = (bf16*)((char*)Xe + ((size_t)64 << 20)); // reuse: adjacency (32 MB)
    bf16*  Xc   = xnB;                                    // reuse: tanh(A@Xe/k)

    const int T = 256;
    auto g1 = [&](size_t n) { return (unsigned)((n + T - 1) / T); };

    // 1. precision conversion / transposes
    cvt_f32_bf16     <<<g1((size_t)M * K1),  T, 0, stream>>>(X,  Xb,  (size_t)M * K1);
    cvt_transpose_pad<<<g1((size_t)D * K1),  T, 0, stream>>>(Wb, WbT, K1, D,  D);
    cvt_transpose_pad<<<g1((size_t)NCp * D), T, 0, stream>>>(Wm, WmT, D,  NC, NCp);
    cvt_transpose_pad<<<g1((size_t)NCp * D), T, 0, stream>>>(Wc, WcT, D,  NC, NCp);

    // 2. Xe = X @ W_buffer          [4096 x 8192], K=512
    wmma_gemm_bf16<0><<<dim3(D / 256, M / 128), T, 0, stream>>>(
        Xb, WbT, Xe, nullptr, M, D, K1, D, nullptr);

    // 3. row norms + fused prep (relu/Xn/Xe^T in bf16)
    row_rnorm<<<M, T, 0, stream>>>(Xe, rn, D);
    prep_pass<<<g1((size_t)M * D), T, 0, stream>>>(Xe, rn, reluB, xnB, xeT, M, D);

    // 4. sim = Xn @ Xn^T            [4096 x 4096], K=8192  (overwrites Xe region)
    wmma_gemm_bf16<0><<<dim3(M / 256, M / 128), T, 0, stream>>>(
        xnB, xnB, Smat, nullptr, M, M, D, M, nullptr);

    // 5. exact top-k -> bf16 adjacency
    topk_adjacency<<<M, T, 0, stream>>>(Smat, Adj, M, kp);

    // 6. Xc = tanh((Adj @ Xe) / k)  [4096 x 8192], K=4096  (bf16 out, reuses Xn slot)
    wmma_gemm_bf16<1><<<dim3(D / 256, M / 128), T, 0, stream>>>(
        Adj, xeT, nullptr, Xc, M, D, M, D, kp);

    // 7. out  = relu(Xe) @ W_main   [4096 x 1000], K=8192
    wmma_gemm_bf16<2><<<dim3(NCp / 256, M / 128), T, 0, stream>>>(
        reluB, WmT, out, nullptr, M, NCp, D, NC, nullptr);

    // 8. out += Xc @ W_comp         (C_COEF = 1.0)
    wmma_gemm_bf16<3><<<dim3(NCp / 256, M / 128), T, 0, stream>>>(
        Xc, WcT, out, nullptr, M, NCp, D, NC, nullptr);
}
